// Attention_gate_21457656611273
// MI455X (gfx1250) — matrix operand order repaired and checked by reading, not compiled, not run on hardware
//
#include <hip/hip_runtime.h>

// ---------------------------------------------------------------------------
// CDNA5 / gfx1250 fused kernel for: subsample -> 3x(proj+BN+ReLU) -> 4x4
// attention -> proj back -> BN -> residual add (broadcast over w).
// GEMMs use v_wmma_f32_16x16x32_f16 (f16 A/B, f32 accumulate).
// ---------------------------------------------------------------------------

typedef __attribute__((ext_vector_type(16))) _Float16 v16h;
typedef __attribute__((ext_vector_type(8)))  float    v8f;

union Frag { uint4 q[2]; v16h v; unsigned u[8]; };

#define EPSV 1e-5f

// Problem sizes (fixed by the reference)
static constexpr int N_TOT = 2048;
static constexpr int C_TOT = 512;
static constexpr int D_TOT = 64;

// Workspace layout (bytes)
static constexpr size_t OFF_WSW123 = 0;              // [3][4 dt][16 s][32 lane][8] u32 (f16 pairs)
static constexpr size_t OFF_WSW4   = 196608;         // [32 ct][2 s][32 lane][8] u32
static constexpr size_t OFF_A123   = 262144;         // alpha123: 192 f32
static constexpr size_t OFF_B123   = OFF_A123 + 192 * 4;
static constexpr size_t OFF_A4     = OFF_B123 + 192 * 4;  // alpha4: 512 f32
static constexpr size_t OFF_B4     = OFF_A4 + 512 * 4;
static constexpr size_t OFF_ATT    = 268288;         // att f16 [8192 rows][64 d]

// ---------------------------------------------------------------------------
// Prep: fold BN into (alpha, beta); swizzle weights (f32 -> f16) into exact
// WMMA B-fragment order. B-fragment layout assumption (from ISA 7.12.5):
//   lane L holds column n = ct*16 + (L&15); k-off = 8*(L>>4);
//   u32 v (0..7) packs B[kbase + koff + 2v + 8*(v>>2)][n], B[... + 1][n].
// Stored per-fragment as [lane][8 u32] so each lane loads 2x b128 coalesced.
// ---------------------------------------------------------------------------
__global__ __launch_bounds__(256) void prep_kernel(
    const float* __restrict__ W123, const float* __restrict__ b123,
    const float* __restrict__ g123, const float* __restrict__ be123,
    const float* __restrict__ m123, const float* __restrict__ v123,
    const float* __restrict__ W4,   const float* __restrict__ b4,
    const float* __restrict__ g4,   const float* __restrict__ be4,
    const float* __restrict__ m4,   const float* __restrict__ v4,
    unsigned* __restrict__ wsw123, unsigned* __restrict__ wsw4,
    float* __restrict__ alpha123, float* __restrict__ beta123,
    float* __restrict__ alpha4,   float* __restrict__ beta4)
{
    const int W123SZ = 3 * 4 * 16 * 32 * 8;   // 49152
    const int W4SZ   = 32 * 2 * 32 * 8;       // 16384
    int tid = blockIdx.x * blockDim.x + threadIdx.x;

    if (tid < W123SZ) {
        int v    = tid & 7;
        int lane = (tid >> 3) & 31;
        int s    = (tid >> 8) & 15;
        int ct   = (tid >> 12) & 3;
        int kp   = tid >> 14;
        int d    = ct * 16 + (lane & 15);
        int c    = 32 * s + 8 * (lane >> 4) + 2 * v + 8 * (v >> 2);
        union { unsigned u; _Float16 h[2]; } p;
        p.h[0] = (_Float16)W123[(kp * 64 + d) * 512 + c];
        p.h[1] = (_Float16)W123[(kp * 64 + d) * 512 + c + 1];
        wsw123[tid] = p.u;
    } else if (tid < W123SZ + W4SZ) {
        int t    = tid - W123SZ;
        int v    = t & 7;
        int lane = (t >> 3) & 31;
        int s    = (t >> 8) & 1;
        int ct   = t >> 9;
        int c    = ct * 16 + (lane & 15);
        int d    = 32 * s + 8 * (lane >> 4) + 2 * v + 8 * (v >> 2);
        union { unsigned u; _Float16 h[2]; } p;
        p.h[0] = (_Float16)W4[c * 64 + d];
        p.h[1] = (_Float16)W4[c * 64 + d + 1];
        wsw4[t] = p.u;
    } else if (tid < W123SZ + W4SZ + 192) {
        int t = tid - (W123SZ + W4SZ);        // t = kp*64 + d
        float a = g123[t] * rsqrtf(v123[t] + EPSV);
        alpha123[t] = a;
        beta123[t]  = (b123[t] - m123[t]) * a + be123[t];
    } else if (tid < W123SZ + W4SZ + 192 + 512) {
        int t = tid - (W123SZ + W4SZ + 192);  // t = c
        float a = g4[t] * rsqrtf(v4[t] + EPSV);
        alpha4[t] = a;
        beta4[t]  = (b4[t] - m4[t]) * a + be4[t];
    }
}

// ---------------------------------------------------------------------------
// Stage A: per block, 8 samples (32 M-rows of (n,l)).  6 waves = 192 threads:
// wave = (kp in 0..2) * 2 + (ms in 0..1).  Each wave: 16x16 M-tile x full
// D=64 (4 accumulators), K=512 in 16 WMMA steps.  Then in-block 4x4 attention.
// LDS A layout: f16, row stride 268 dwords (16B-aligned b128 fragment reads,
// bank-spread).  A-fragment (16-bit A 16x32, ISA 7.12.2): lane L -> row L&15,
// half = L>>4; u32 pair index within kstep = 4*half + v (+4 for v>=4).
// ---------------------------------------------------------------------------
static constexpr int ASTR_U32 = 268;           // dwords per LDS A row
static constexpr int ASTR_H   = ASTR_U32 * 2;  // halves per LDS A row

__global__ __launch_bounds__(192) void stageA_kernel(
    const float* __restrict__ x,
    const unsigned* __restrict__ wsw123,
    const float* __restrict__ alpha123, const float* __restrict__ beta123,
    _Float16* __restrict__ att_h)
{
    __shared__ __attribute__((aligned(16))) _Float16 s_a[32 * ASTR_H]; // 34304 B
    __shared__ __attribute__((aligned(16))) float    s_act[3][32][68]; // 26112 B
    __shared__ float s_p[8][4][4];

    const int tid  = threadIdx.x;
    const int lane = tid & 31;
    const int wave = tid >> 5;
    const int kp   = wave >> 1;      // projection 0..2
    const int ms   = wave & 1;       // M-subtile 0..1
    const int half = lane >> 4;
    const int lrow = lane & 15;
    const int nbase = blockIdx.x * 8;

    // ---- Stage x tile -> LDS f16.  xs positions {0,2,8,10} = (.x,.z) of
    // float4 slots 0 and 2 of each (n,c) HW-plane (one 64B line).
    for (int i = tid; i < 8 * 512; i += 192) {
        int nloc = i >> 9;
        int c    = i & 511;
        const float4* xp = (const float4*)x + ((size_t)(nbase + nloc) * 512 + c) * 4;
        float4 a0 = xp[0];
        float4 a2 = xp[2];
        int rb = nloc * 4;
        s_a[(rb + 0) * ASTR_H + c] = (_Float16)a0.x;
        s_a[(rb + 1) * ASTR_H + c] = (_Float16)a0.z;
        s_a[(rb + 2) * ASTR_H + c] = (_Float16)a2.x;
        s_a[(rb + 3) * ASTR_H + c] = (_Float16)a2.z;
    }
    __syncthreads();

    // ---- WMMA: 16 k-steps x 4 d-tiles
    v8f acc0 = {0.f,0.f,0.f,0.f,0.f,0.f,0.f,0.f};
    v8f acc[4] = {acc0, acc0, acc0, acc0};
    const unsigned* arow = (const unsigned*)s_a + (ms * 16 + lrow) * ASTR_U32;
    #pragma unroll 4
    for (int s = 0; s < 16; ++s) {
        Frag af;
        int u0 = 16 * s + 4 * half;
        af.q[0] = *(const uint4*)(arow + u0);
        af.q[1] = *(const uint4*)(arow + u0 + 8);
        #pragma unroll
        for (int dt = 0; dt < 4; ++dt) {
            const unsigned* bp = wsw123 + ((((kp * 4 + dt) * 16 + s) * 32 + lane) * 8);
            Frag bf;
            bf.q[0] = *(const uint4*)bp;
            bf.q[1] = *(const uint4*)(bp + 4);
            acc[dt] = __builtin_amdgcn_wmma_f32_16x16x32_f16(
                false, af.v, false, bf.v, (short)0, acc[dt], false, false);
        }
    }

    // ---- Epilogue: BN (alpha*acc+beta) + ReLU -> LDS act[kp][row][d]
    #pragma unroll
    for (int dt = 0; dt < 4; ++dt) {
        int d = dt * 16 + lrow;
        float al = alpha123[kp * 64 + d];
        float be = beta123[kp * 64 + d];
        #pragma unroll
        for (int j = 0; j < 8; ++j) {
            int m = 8 * half + j;
            s_act[kp][ms * 16 + m][d] = fmaxf(acc[dt][j] * al + be, 0.0f);
        }
    }
    __syncthreads();

    // ---- Attention: 8 samples x 4x4 scores, softmax over m (4-lane groups)
    if (tid < 128) {
        int m    = tid & 3;
        int l    = (tid >> 2) & 3;
        int nloc = tid >> 4;
        const float4* qr = (const float4*)&s_act[0][nloc * 4 + l][0];
        const float4* kr = (const float4*)&s_act[1][nloc * 4 + m][0];
        float sdot = 0.0f;
        #pragma unroll
        for (int i = 0; i < 16; ++i) {
            float4 a = qr[i], b = kr[i];
            sdot += a.x * b.x + a.y * b.y + a.z * b.z + a.w * b.w;
        }
        float mx = sdot;
        mx = fmaxf(mx, __shfl_xor(mx, 1, 4));
        mx = fmaxf(mx, __shfl_xor(mx, 2, 4));
        float e = __expf(sdot - mx);
        float sum = e;
        sum += __shfl_xor(sum, 1, 4);
        sum += __shfl_xor(sum, 2, 4);
        s_p[nloc][l][m] = e / sum;
    }
    __syncthreads();

    // ---- att[l][d] = sum_m p[l][m] * v[m][d]; write f16 rows for stage 3
    if (tid < 128) {
        int dq   = tid & 3;
        int l    = (tid >> 2) & 3;
        int nloc = tid >> 4;
        float p0 = s_p[nloc][l][0], p1 = s_p[nloc][l][1];
        float p2 = s_p[nloc][l][2], p3 = s_p[nloc][l][3];
        const float* v0 = &s_act[2][nloc * 4 + 0][0];
        const float* v1 = &s_act[2][nloc * 4 + 1][0];
        const float* v2 = &s_act[2][nloc * 4 + 2][0];
        const float* v3 = &s_act[2][nloc * 4 + 3][0];
        _Float16* dst = att_h + (size_t)(blockIdx.x * 32 + nloc * 4 + l) * 64 + dq * 16;
        #pragma unroll
        for (int dd = 0; dd < 16; ++dd) {
            int d = dq * 16 + dd;
            dst[dd] = (_Float16)(p0 * v0[d] + p1 * v1[d] + p2 * v2[d] + p3 * v3[d]);
        }
    }
}

// ---------------------------------------------------------------------------
// Final: out2 = att(8192x64) @ W4^T(64x512) via WMMA (2 k-steps), then
// BN2 and residual: out[n,c,h,0..3] = x[n,c,h,0..3] + bn2[n,c,l=h]
// (each C/D element maps to exactly one contiguous float4 of x/out).
// Block = 256 threads (8 waves): wave -> (ms 0..3, cg 0..1); 16 n per block.
// ---------------------------------------------------------------------------
__global__ __launch_bounds__(256) void final_kernel(
    const float* __restrict__ x,
    const _Float16* __restrict__ att_h,
    const unsigned* __restrict__ wsw4,
    const float* __restrict__ alpha4, const float* __restrict__ beta4,
    float* __restrict__ out)
{
    const int tid  = threadIdx.x;
    const int lane = tid & 31;
    const int wave = tid >> 5;
    const int ms   = wave & 3;
    const int cg   = wave >> 2;
    const int half = lane >> 4;
    const int lrow = lane & 15;
    const int rowbase = blockIdx.x * 64 + ms * 16;

    // A fragments for both k-steps (K = 64), reused across all c-tiles.
    Frag af[2];
    const unsigned* arow = (const unsigned*)att_h + (size_t)(rowbase + lrow) * 32;
    #pragma unroll
    for (int s = 0; s < 2; ++s) {
        int u0 = 16 * s + 4 * half;
        af[s].q[0] = *(const uint4*)(arow + u0);
        af[s].q[1] = *(const uint4*)(arow + u0 + 8);
    }

    for (int i = 0; i < 16; ++i) {
        int ct = cg * 16 + i;
        v8f acc = {0.f,0.f,0.f,0.f,0.f,0.f,0.f,0.f};
        #pragma unroll
        for (int s = 0; s < 2; ++s) {
            const unsigned* bp = wsw4 + (((ct * 2 + s) * 32 + lane) * 8);
            Frag bf;
            bf.q[0] = *(const uint4*)bp;
            bf.q[1] = *(const uint4*)(bp + 4);
            acc = __builtin_amdgcn_wmma_f32_16x16x32_f16(
                false, af[s].v, false, bf.v, (short)0, acc, false, false);
        }
        int c = ct * 16 + lrow;
        float al = alpha4[c];
        float be = beta4[c];
        #pragma unroll
        for (int j = 0; j < 8; ++j) {
            int row = rowbase + 8 * half + j;   // global (n,l) row
            int n = row >> 2;
            int l = row & 3;
            float bn2 = acc[j] * al + be;
            size_t q4 = ((size_t)n * 512 + c) * 4 + (size_t)l;
            float4 xv = ((const float4*)x)[q4];
            float4 ov = make_float4(xv.x + bn2, xv.y + bn2, xv.z + bn2, xv.w + bn2);
            ((float4*)out)[q4] = ov;
        }
    }
}

// ---------------------------------------------------------------------------
extern "C" void kernel_launch(void* const* d_in, const int* in_sizes, int n_in,
                              void* d_out, int out_size, void* d_ws, size_t ws_size,
                              hipStream_t stream)
{
    (void)in_sizes; (void)n_in; (void)out_size; (void)ws_size;
    const float* x     = (const float*)d_in[0];
    const float* W123  = (const float*)d_in[1];
    const float* b123  = (const float*)d_in[2];
    const float* g123  = (const float*)d_in[3];
    const float* be123 = (const float*)d_in[4];
    const float* m123  = (const float*)d_in[5];
    const float* v123  = (const float*)d_in[6];
    const float* W4    = (const float*)d_in[7];
    const float* b4    = (const float*)d_in[8];
    const float* g4    = (const float*)d_in[9];
    const float* be4   = (const float*)d_in[10];
    const float* m4    = (const float*)d_in[11];
    const float* v4    = (const float*)d_in[12];
    float* out = (float*)d_out;

    char* ws = (char*)d_ws;
    unsigned* wsw123  = (unsigned*)(ws + OFF_WSW123);
    unsigned* wsw4    = (unsigned*)(ws + OFF_WSW4);
    float*    alpha123= (float*)(ws + OFF_A123);
    float*    beta123 = (float*)(ws + OFF_B123);
    float*    alpha4  = (float*)(ws + OFF_A4);
    float*    beta4   = (float*)(ws + OFF_B4);
    _Float16* att_h   = (_Float16*)(ws + OFF_ATT);

    const int prep_work = 3*4*16*32*8 + 32*2*32*8 + 192 + 512;
    prep_kernel<<<(prep_work + 255) / 256, 256, 0, stream>>>(
        W123, b123, g123, be123, m123, v123, W4, b4, g4, be4, m4, v4,
        wsw123, wsw4, alpha123, beta123, alpha4, beta4);

    stageA_kernel<<<N_TOT / 8, 192, 0, stream>>>(
        x, wsw123, alpha123, beta123, att_h);

    final_kernel<<<N_TOT / 16, 256, 0, stream>>>(
        x, att_h, wsw4, alpha4, beta4, out);
}
